// Branch1234_56427280335365
// MI455X (gfx1250) — compile-verified, matrix operand order checked by reading, not run on hardware
//
#include <hip/hip_runtime.h>
#include <hip/hip_bf16.h>
#include <math.h>

// ---------------------------------------------------------------------------
// Types for CDNA5 WMMA
// ---------------------------------------------------------------------------
typedef __attribute__((ext_vector_type(16))) __bf16 v16bf;
typedef __attribute__((ext_vector_type(8)))  float  v8f;

struct alignas(16) U128 { unsigned int x, y, z, w; };
struct alignas(8)  U64p { unsigned int lo, hi; };
union FragAB { U128 u[2]; v16bf v; };   // 32 bytes: 16 bf16 = one WMMA A/B fragment

__device__ __forceinline__ unsigned short f2bf(float f) {
  // round-to-nearest-even fp32 -> bf16
  unsigned int u = __float_as_uint(f);
  unsigned int r = u + 0x7FFFu + ((u >> 16) & 1u);
  return (unsigned short)(r >> 16);
}
__device__ __forceinline__ unsigned int f2bf2(float lo, float hi) {
  return (unsigned int)f2bf(lo) | ((unsigned int)f2bf(hi) << 16);
}

__device__ __forceinline__ float siluf(float x) { return x / (1.f + __expf(-x)); }

// ---------------------------------------------------------------------------
// Generic bf16-WMMA GEMM:
//   Out[b, tok, o] (+)= bias[o] + sum_k Act[b, tok, k] * W[o, k]
// Act addressed with arbitrary element strides. W row-major (O x Kfull).
// fp32 accumulate. 256 threads / 8 waves; macro tile 128 tok x 64 out;
// 2x2 register tile per wave -> 4 v_wmma per K-chunk per wave.
// Staging (block-uniform paths):
//   - float4 vector path when k is contiguous and row stride % 4 == 0
//   - scalar path with lane mapping chosen by stride pattern (lanes along the
//     contiguous dimension -> coalesced for token-major AND channel-major A)
//   - predicated edge path
// LDS pitch padded to 40 halfwords (80 B): the 16 b128 fragment-row starts
// land on 20*m mod 64 -> all distinct banks, conflict-free ds_load_b128.
// ---------------------------------------------------------------------------
#define TM  128
#define TO  64
#define TK  32
#define TKP 40   // padded LDS pitch (halfwords); 80 B keeps 16 B alignment

__global__ __launch_bounds__(256)
void gemm_wmma_bf16(const float* __restrict__ A,
                    long long aRow, long long aCol, long long aBatch,
                    const float* __restrict__ Wt, int Kfull,
                    const float* __restrict__ bias,
                    float* __restrict__ Out,
                    long long oRow, long long oCol, long long oBatch,
                    int M, int O, int K, int accumulate)
{
  __shared__ unsigned short lsA[TM * TKP];   // 10 KB
  __shared__ unsigned short lsB[TO * TKP];   //  5 KB

  const int tid  = threadIdx.x;
  const int wave = tid >> 5;          // wave32
  const int lane = tid & 31;
  const int wr   = wave & 3;          // token sub-tile (4 x 32 rows)
  const int wc   = wave >> 2;         // out sub-tile   (2 x 32 cols)
  const int tokBase = blockIdx.x * TM;
  const int oBase   = blockIdx.y * TO;
  const int b       = blockIdx.z;

  const float* Ab = A   + (long long)b * aBatch;
  float*       Ob = Out + (long long)b * oBatch;

  v8f acc00 = {}, acc01 = {}, acc10 = {}, acc11 = {};

  const int kb8  = 8 * (lane >> 4);   // K-half base within fragment (0 or 8)
  const int mloc = lane & 15;

  const int sT = tid >> 5;            // k-major staging: row 0..7 (step 8)
  const int sK = tid & 31;            // k-major staging: k within chunk

  const bool aVec = (aCol == 1) && ((aRow & 3LL) == 0);
  const bool wVec = ((Kfull & 3) == 0);

  for (int k0 = 0; k0 < K; k0 += TK) {
    const bool fullK = (k0 + TK <= K);
    const bool fullA = fullK && (tokBase + TM <= M);
    const bool fullB = fullK && (oBase + TO <= O);

    // ---- stage A tile (128 x 32) fp32 -> bf16 ----
    if (fullA && aVec) {
      // vector path: each thread 4x float4 (k contiguous)
#pragma unroll
      for (int j = 0; j < 4; ++j) {
        int e4 = tid + j * 256;
        int t = e4 >> 3, kq = e4 & 7;
        const float* pr = Ab + (long long)(tokBase + t) * aRow + (k0 + 4 * kq);
        float4 v = *(const float4*)pr;
        U64p pk; pk.lo = f2bf2(v.x, v.y); pk.hi = f2bf2(v.z, v.w);
        *(U64p*)&lsA[t * TKP + 4 * kq] = pk;
      }
    } else if (fullA) {
      if (aCol == 1) {
        // lanes along k (contiguous k, unaligned rows)
        const float* p = Ab + (long long)(tokBase + sT) * aRow + (k0 + sK);
        float v[16];
#pragma unroll
        for (int j = 0; j < 16; ++j) v[j] = p[(long long)(8 * j) * aRow];
#pragma unroll
        for (int j = 0; j < 16; ++j) lsA[(sT + 8 * j) * TKP + sK] = f2bf(v[j]);
      } else {
        // lanes along tokens (channel-major A: token stride is smallest)
        int t0 = tid & 127, kq0 = (tid >> 7) * 16;
        const float* p = Ab + (long long)(tokBase + t0) * aRow + (long long)(k0 + kq0) * aCol;
        float v[16];
#pragma unroll
        for (int j = 0; j < 16; ++j) v[j] = p[(long long)j * aCol];
#pragma unroll
        for (int j = 0; j < 16; ++j) lsA[t0 * TKP + kq0 + j] = f2bf(v[j]);
      }
    } else {
      // edge tile: predicated scalar
#pragma unroll
      for (int j = 0; j < 16; ++j) {
        int gt = tokBase + sT + 8 * j, gk = k0 + sK;
        float v = (gt < M && gk < K) ? Ab[(long long)gt * aRow + (long long)gk * aCol] : 0.f;
        lsA[(sT + 8 * j) * TKP + sK] = f2bf(v);
      }
    }

    // ---- stage W tile (64 x 32), W row-major O x K ----
    if (fullB && wVec) {
#pragma unroll
      for (int j = 0; j < 2; ++j) {
        int e4 = tid + j * 256;
        int o = e4 >> 3, kq = e4 & 7;
        const float* pr = Wt + (long long)(oBase + o) * Kfull + (k0 + 4 * kq);
        float4 v = *(const float4*)pr;
        U64p pk; pk.lo = f2bf2(v.x, v.y); pk.hi = f2bf2(v.z, v.w);
        *(U64p*)&lsB[o * TKP + 4 * kq] = pk;
      }
    } else if (fullB) {
      const float* p = Wt + (long long)(oBase + sT) * Kfull + (k0 + sK);
      float v[8];
#pragma unroll
      for (int j = 0; j < 8; ++j) v[j] = p[(long long)(8 * j) * Kfull];
#pragma unroll
      for (int j = 0; j < 8; ++j) lsB[(sT + 8 * j) * TKP + sK] = f2bf(v[j]);
    } else {
#pragma unroll
      for (int j = 0; j < 8; ++j) {
        int go = oBase + sT + 8 * j, gk = k0 + sK;
        float v = (go < O && gk < K) ? Wt[(long long)go * Kfull + gk] : 0.f;
        lsB[(sT + 8 * j) * TKP + sK] = f2bf(v);
      }
    }

    // prefetch next W chunk into GL2 (gfx1250 global_prefetch_b8)
    if (k0 + TK < K && tid == 0)
      __builtin_prefetch(&Wt[(long long)oBase * Kfull + k0 + TK], 0, 1);
    __syncthreads();

    // ---- fragments: two b128 LDS reads each (ISA 16-bit A layout) ----
    FragAB a0, a1, b0, b1;
    int ar0 = (32 * wr + mloc) * TKP;
    int ar1 = ar0 + 16 * TKP;
    a0.u[0] = *(const U128*)&lsA[ar0 + kb8];
    a0.u[1] = *(const U128*)&lsA[ar0 + kb8 + 16];
    a1.u[0] = *(const U128*)&lsA[ar1 + kb8];
    a1.u[1] = *(const U128*)&lsA[ar1 + kb8 + 16];
    int br0 = (32 * wc + mloc) * TKP;
    int br1 = br0 + 16 * TKP;
    b0.u[0] = *(const U128*)&lsB[br0 + kb8];
    b0.u[1] = *(const U128*)&lsB[br0 + kb8 + 16];
    b1.u[0] = *(const U128*)&lsB[br1 + kb8];
    b1.u[1] = *(const U128*)&lsB[br1 + kb8 + 16];

    acc00 = __builtin_amdgcn_wmma_f32_16x16x32_bf16(false, a0.v, false, b0.v,
                                                    (short)0, acc00, false, false);
    acc01 = __builtin_amdgcn_wmma_f32_16x16x32_bf16(false, a0.v, false, b1.v,
                                                    (short)0, acc01, false, false);
    acc10 = __builtin_amdgcn_wmma_f32_16x16x32_bf16(false, a1.v, false, b0.v,
                                                    (short)0, acc10, false, false);
    acc11 = __builtin_amdgcn_wmma_f32_16x16x32_bf16(false, a1.v, false, b1.v,
                                                    (short)0, acc11, false, false);
    __syncthreads();
  }

  // ---- epilogue: C layout lane -> N = lane&15, VGPR v -> M = v + 8*(lane>>4)
  const int mhalf = 8 * (lane >> 4);
  const int n0 = oBase + 32 * wc + mloc;
  const int n1 = n0 + 16;
  const float bv0 = (bias && n0 < O) ? bias[n0] : 0.f;
  const float bv1 = (bias && n1 < O) ? bias[n1] : 0.f;
#pragma unroll
  for (int ti = 0; ti < 2; ++ti) {
    const v8f& c0 = ti ? acc10 : acc00;
    const v8f& c1 = ti ? acc11 : acc01;
#pragma unroll
    for (int v = 0; v < 8; ++v) {
      int tok = tokBase + 32 * wr + 16 * ti + v + mhalf;
      if (tok >= M) continue;
      long long base = (long long)tok * oRow;
      if (n0 < O) {
        long long idx = base + (long long)n0 * oCol;
        float r = c0[v] + bv0;
        Ob[idx] = accumulate ? (Ob[idx] + r) : r;
      }
      if (n1 < O) {
        long long idx = base + (long long)n1 * oCol;
        float r = c1[v] + bv1;
        Ob[idx] = accumulate ? (Ob[idx] + r) : r;
      }
    }
  }
}

// ---------------------------------------------------------------------------
// Depthwise causal conv (K=4) + SiLU.  Input xi = xz[..., :di] (row stride xzs)
// ---------------------------------------------------------------------------
__global__ void dwconv_silu(const float* __restrict__ xz, int xzs,
                            const float* __restrict__ w, const float* __restrict__ cb,
                            float* __restrict__ out, int Bn, int L, int di)
{
  long long idx = (long long)blockIdx.x * blockDim.x + threadIdx.x;
  long long tot = (long long)Bn * L * di;
  if (idx >= tot) return;
  int d = (int)(idx % di);
  int l = (int)((idx / di) % L);
  int b = (int)(idx / ((long long)di * L));
  float acc = cb[d];
#pragma unroll
  for (int k = 0; k < 4; ++k) {
    int ls = l - 3 + k;
    if (ls >= 0)
      acc += w[d * 4 + k] * xz[((long long)b * L + ls) * xzs + d];
  }
  out[((long long)b * L + l) * di + d] = siluf(acc);
}

// ---------------------------------------------------------------------------
// Fused selective scan: softplus(dt), A=-exp(A_log), h recurrence with N=16
// states in registers, +D*x skip, *silu(z) gate.  One thread per (b, d).
// y aliases the dead xi-half of xz (stride ys).
// ---------------------------------------------------------------------------
__global__ void selective_scan(const float* __restrict__ xz, int xzs,
                               const float* __restrict__ xi,
                               const float* __restrict__ xdbl, int xdw, int dt_rank,
                               const float* __restrict__ dtp,
                               const float* __restrict__ A_log,
                               const float* __restrict__ Dp,
                               const float* __restrict__ dt_b,
                               float* __restrict__ y, int ys,
                               int Bn, int L, int di)
{
  int idx = blockIdx.x * blockDim.x + threadIdx.x;
  if (idx >= Bn * di) return;
  int b = idx / di, d = idx % di;

  float a[16], h[16];
#pragma unroll
  for (int n = 0; n < 16; ++n) { a[n] = -__expf(A_log[d * 16 + n]); h[n] = 0.f; }
  const float Dd  = Dp[d];
  const float dtb = dt_b[d];

  for (int l = 0; l < L; ++l) {
    long long row = (long long)b * L + l;
    float dtv = dtp[row * di + d] + dtb;
    dtv = (dtv > 20.f) ? dtv : log1pf(__expf(dtv));        // softplus
    float xv = xi[row * di + d];
    float zv = xz[row * xzs + di + d];
    const float* xd = xdbl + row * xdw + dt_rank;
    float dx = dtv * xv;
    float accy = 0.f;
#pragma unroll
    for (int n = 0; n < 16; ++n) {
      float Bv = xd[n];
      float Cv = xd[16 + n];
      h[n] = __expf(dtv * a[n]) * h[n] + dx * Bv;
      accy += h[n] * Cv;
    }
    float yo = accy + Dd * xv;
    y[row * ys + d] = yo * siluf(zv);
  }
}

// ---------------------------------------------------------------------------
// LayerNorm over rows (one wave32 per row, shuffle reduction)
// ---------------------------------------------------------------------------
__global__ void layernorm_rows(float* __restrict__ t,
                               const float* __restrict__ w, const float* __restrict__ bias,
                               long long rows, int Dn)
{
  long long wid = ((long long)blockIdx.x * blockDim.x + threadIdx.x) >> 5;
  int lane = threadIdx.x & 31;
  if (wid >= rows) return;
  float* r = t + wid * Dn;
  float s = 0.f, ss = 0.f;
  for (int i = lane; i < Dn; i += 32) { float v = r[i]; s += v; ss += v * v; }
#pragma unroll
  for (int off = 16; off >= 1; off >>= 1) {
    s  += __shfl_xor(s,  off, 32);
    ss += __shfl_xor(ss, off, 32);
  }
  float mu  = s / Dn;
  float var = ss / Dn - mu * mu;
  float inv = rsqrtf(var + 1e-5f);
  for (int i = lane; i < Dn; i += 32)
    r[i] = (r[i] - mu) * inv * w[i] + bias[i];
}

// ---------------------------------------------------------------------------
// Align-corners bilinear resize (B,C,H,W) -> (B,C,OH,OW)
// ---------------------------------------------------------------------------
__global__ void resize_bilinear(const float* __restrict__ in, float* __restrict__ out,
                                int Bn, int C, int H, int W, int OH, int OW)
{
  long long idx = (long long)blockIdx.x * blockDim.x + threadIdx.x;
  long long tot = (long long)Bn * C * OH * OW;
  if (idx >= tot) return;
  int ox = (int)(idx % OW);
  int oy = (int)((idx / OW) % OH);
  long long bc = idx / ((long long)OW * OH);   // b*C + c
  float fy = (OH > 1) ? oy * (float)(H - 1) / (float)(OH - 1) : 0.f;
  float fx = (OW > 1) ? ox * (float)(W - 1) / (float)(OW - 1) : 0.f;
  int y0 = (int)floorf(fy); int y1 = min(y0 + 1, H - 1); float wy = fy - y0;
  int x0 = (int)floorf(fx); int x1 = min(x0 + 1, W - 1); float wx = fx - x0;
  const float* p = in + bc * H * W;
  float v00 = p[y0 * W + x0], v01 = p[y0 * W + x1];
  float v10 = p[y1 * W + x0], v11 = p[y1 * W + x1];
  float top = v00 * (1.f - wx) + v01 * wx;
  float bot = v10 * (1.f - wx) + v11 * wx;
  out[idx] = top * (1.f - wy) + bot * wy;
}

// ---------------------------------------------------------------------------
// out[b,o,hw] += t[b, hw, o]   (x2 branch accumulation)
// ---------------------------------------------------------------------------
__global__ void add_transpose(float* __restrict__ out, const float* __restrict__ t,
                              int Bn, int O, int HW)
{
  long long idx = (long long)blockIdx.x * blockDim.x + threadIdx.x;
  long long tot = (long long)Bn * O * HW;
  if (idx >= tot) return;
  int hw = (int)(idx % HW);
  int o  = (int)((idx / HW) % O);
  int b  = (int)(idx / ((long long)HW * O));
  out[idx] += t[((long long)b * HW + hw) * O + o];
}

// ---------------------------------------------------------------------------
// Host side
// ---------------------------------------------------------------------------
static inline int cdiv(long long a, long long b) { return (int)((a + b - 1) / b); }

struct MP {
  const float *in_w, *conv_w, *conv_b, *xproj_w, *dt_w, *dt_b, *A_log, *D, *out_w;
};

static void launch_gemm(hipStream_t s,
                        const float* A, long long aRow, long long aCol, long long aBatch,
                        const float* Wt, int Kfull, const float* bias,
                        float* Out, long long oRow, long long oCol, long long oBatch,
                        int M, int O, int K, int Bn, int acc)
{
  dim3 g(cdiv(M, TM), cdiv(O, TO), Bn);
  gemm_wmma_bf16<<<g, 256, 0, s>>>(A, aRow, aCol, aBatch, Wt, Kfull, bias,
                                   Out, oRow, oCol, oBatch, M, O, K, acc);
}

extern "C" void kernel_launch(void* const* d_in, const int* in_sizes, int n_in,
                              void* d_out, int out_size, void* d_ws, size_t ws_size,
                              hipStream_t stream)
{
  (void)in_sizes; (void)n_in; (void)out_size; (void)ws_size;

  const int B = 4, C = 96, H = 96, Wd = 96, HW = H * Wd, OUT = 96;
  const int SIZE = 16, S2 = SIZE * SIZE;             // 256
  const int L1 = HW;                                 // 9216 tokens (m1/m2)

  const float* x = (const float*)d_in[0];
  MP m1 = { (const float*)d_in[1],  (const float*)d_in[2],  (const float*)d_in[3],
            (const float*)d_in[4],  (const float*)d_in[5],  (const float*)d_in[6],
            (const float*)d_in[7],  (const float*)d_in[8],  (const float*)d_in[9] };
  MP m2 = { (const float*)d_in[10], (const float*)d_in[11], (const float*)d_in[12],
            (const float*)d_in[13], (const float*)d_in[14], (const float*)d_in[15],
            (const float*)d_in[16], (const float*)d_in[17], (const float*)d_in[18] };
  MP m3 = { (const float*)d_in[19], (const float*)d_in[20], (const float*)d_in[21],
            (const float*)d_in[22], (const float*)d_in[23], (const float*)d_in[24],
            (const float*)d_in[25], (const float*)d_in[26], (const float*)d_in[27] };
  const float* n1_w = (const float*)d_in[28]; const float* n1_b = (const float*)d_in[29];
  const float* n2_w = (const float*)d_in[30]; const float* n2_b = (const float*)d_in[31];
  const float* n3_w = (const float*)d_in[32]; const float* n3_b = (const float*)d_in[33];
  const float* lc_w = (const float*)d_in[34]; const float* lc_b = (const float*)d_in[35];
  const float* spa_lin_w = (const float*)d_in[36]; const float* spa_lin_b = (const float*)d_in[37];
  const float* lsp_ch_w  = (const float*)d_in[38]; const float* lsp_ch_b  = (const float*)d_in[39];
  const float* sspa_ch_w = (const float*)d_in[40]; const float* sspa_ch_b = (const float*)d_in[41];

  float* out = (float*)d_out;

  // ---- workspace bump allocator (fp32 elements) ----
  float* ws = (float*)d_ws;
  size_t off = 0;
  auto alloc = [&](size_t n) { float* p = ws + off; off += n; return p; };
  float* XZ   = alloc((size_t)B * L1 * 384);   // m1/m2 xz (m3 fits inside)
  float* CONV = alloc((size_t)B * L1 * 192);
  float* XDBL = alloc((size_t)B * L1 * 38);
  float* DT   = alloc((size_t)B * L1 * 192);
  float* TA   = alloc((size_t)B * L1 * OUT);
  float* TB   = alloc((size_t)B * L1 * OUT);
  float* S3D  = alloc((size_t)B * C * S2);
  float* SOUT = alloc((size_t)B * C * S2);
  float* SLIN = alloc((size_t)B * C * S2);
  float* UP1  = alloc((size_t)B * C * HW);
  float* UP2  = alloc((size_t)B * C * HW);

  const int T = 256;
  auto blocks = [&](long long n) { return cdiv(n, T); };

  // =========================== Branch 1: x1 ==============================
  // out[b,o,hw] = sum_c x[b,c,hw]*lc_w[o,c] + lc_b[o]    (writes all of out)
  launch_gemm(stream, x, /*aRow=*/1, /*aCol=*/HW, /*aBatch=*/(long long)C * HW,
              lc_w, C, lc_b,
              out, /*oRow=*/1, /*oCol=*/HW, /*oBatch=*/(long long)OUT * HW,
              HW, OUT, C, B, /*acc=*/0);

  // =========================== Branch 2: x2 ==============================
  // ---- mamba m1 (tokens = transpose of x, via strides) ----
  {
    int D = 96, di = 192, xzs = 384, dtr = 6, xdw = 38;
    launch_gemm(stream, x, 1, HW, (long long)C * HW, m1.in_w, D, nullptr,
                XZ, xzs, 1, (long long)L1 * xzs, L1, 2 * di, D, B, 0);
    dwconv_silu<<<blocks((long long)B * L1 * di), T, 0, stream>>>(
        XZ, xzs, m1.conv_w, m1.conv_b, CONV, B, L1, di);
    launch_gemm(stream, CONV, di, 1, (long long)L1 * di, m1.xproj_w, di, nullptr,
                XDBL, xdw, 1, (long long)L1 * xdw, L1, xdw, di, B, 0);
    launch_gemm(stream, XDBL, xdw, 1, (long long)L1 * xdw, m1.dt_w, dtr, nullptr,
                DT, di, 1, (long long)L1 * di, L1, di, dtr, B, 0);
    selective_scan<<<blocks(B * di), T, 0, stream>>>(
        XZ, xzs, CONV, XDBL, xdw, dtr, DT, m1.A_log, m1.D, m1.dt_b,
        /*y aliases xi-half of XZ*/ XZ, xzs, B, L1, di);
    launch_gemm(stream, XZ, xzs, 1, (long long)L1 * xzs, m1.out_w, di, nullptr,
                TA, OUT, 1, (long long)L1 * OUT, L1, OUT, di, B, 0);
    layernorm_rows<<<cdiv((long long)B * L1 * 32, T), T, 0, stream>>>(
        TA, n1_w, n1_b, (long long)B * L1, OUT);
  }
  // ---- mamba m2 (input TA token-major) ----
  {
    int D = 96, di = 192, xzs = 384, dtr = 6, xdw = 38;
    launch_gemm(stream, TA, D, 1, (long long)L1 * D, m2.in_w, D, nullptr,
                XZ, xzs, 1, (long long)L1 * xzs, L1, 2 * di, D, B, 0);
    dwconv_silu<<<blocks((long long)B * L1 * di), T, 0, stream>>>(
        XZ, xzs, m2.conv_w, m2.conv_b, CONV, B, L1, di);
    launch_gemm(stream, CONV, di, 1, (long long)L1 * di, m2.xproj_w, di, nullptr,
                XDBL, xdw, 1, (long long)L1 * xdw, L1, xdw, di, B, 0);
    launch_gemm(stream, XDBL, xdw, 1, (long long)L1 * xdw, m2.dt_w, dtr, nullptr,
                DT, di, 1, (long long)L1 * di, L1, di, dtr, B, 0);
    selective_scan<<<blocks(B * di), T, 0, stream>>>(
        XZ, xzs, CONV, XDBL, xdw, dtr, DT, m2.A_log, m2.D, m2.dt_b,
        XZ, xzs, B, L1, di);
    launch_gemm(stream, XZ, xzs, 1, (long long)L1 * xzs, m2.out_w, di, nullptr,
                TB, OUT, 1, (long long)L1 * OUT, L1, OUT, di, B, 0);
    layernorm_rows<<<cdiv((long long)B * L1 * 32, T), T, 0, stream>>>(
        TB, n2_w, n2_b, (long long)B * L1, OUT);
  }
  add_transpose<<<blocks((long long)B * OUT * HW), T, 0, stream>>>(out, TB, B, OUT, HW);

  // =========================== Branch 3: x3 ==============================
  resize_bilinear<<<blocks((long long)B * C * S2), T, 0, stream>>>(
      x, S3D, B, C, H, Wd, SIZE, SIZE);
  {
    int D = 256, di = 512, xzs = 1024, dtr = 16, xdw = 48, L = C;  // L=96 tokens
    launch_gemm(stream, S3D, D, 1, (long long)L * D, m3.in_w, D, nullptr,
                XZ, xzs, 1, (long long)L * xzs, L, 2 * di, D, B, 0);
    dwconv_silu<<<blocks((long long)B * L * di), T, 0, stream>>>(
        XZ, xzs, m3.conv_w, m3.conv_b, CONV, B, L, di);
    launch_gemm(stream, CONV, di, 1, (long long)L * di, m3.xproj_w, di, nullptr,
                XDBL, xdw, 1, (long long)L * xdw, L, xdw, di, B, 0);
    launch_gemm(stream, XDBL, xdw, 1, (long long)L * xdw, m3.dt_w, dtr, nullptr,
                DT, di, 1, (long long)L * di, L, di, dtr, B, 0);
    selective_scan<<<blocks(B * di), T, 0, stream>>>(
        XZ, xzs, CONV, XDBL, xdw, dtr, DT, m3.A_log, m3.D, m3.dt_b,
        XZ, xzs, B, L, di);
    launch_gemm(stream, XZ, xzs, 1, (long long)L * xzs, m3.out_w, di, nullptr,
                SOUT, S2, 1, (long long)L * S2, L, S2, di, B, 0);
    layernorm_rows<<<cdiv((long long)B * L * 32, T), T, 0, stream>>>(
        SOUT, n3_w, n3_b, (long long)B * L, S2);
  }
  resize_bilinear<<<blocks((long long)B * C * HW), T, 0, stream>>>(
      SOUT, UP1, B, C, SIZE, SIZE, H, Wd);
  launch_gemm(stream, UP1, 1, HW, (long long)C * HW, sspa_ch_w, C, sspa_ch_b,
              out, 1, HW, (long long)OUT * HW, HW, OUT, C, B, /*acc=*/1);

  // =========================== Branch 4: x4 ==============================
  launch_gemm(stream, S3D, S2, 1, (long long)C * S2, spa_lin_w, S2, spa_lin_b,
              SLIN, S2, 1, (long long)C * S2, C, S2, S2, B, 0);
  resize_bilinear<<<blocks((long long)B * C * HW), T, 0, stream>>>(
      SLIN, UP2, B, C, SIZE, SIZE, H, Wd);
  launch_gemm(stream, UP2, 1, HW, (long long)C * HW, lsp_ch_w, C, lsp_ch_b,
              out, 1, HW, (long long)OUT * HW, HW, OUT, C, B, /*acc=*/1);
}
